// Encoder_80616536146562
// MI455X (gfx1250) — matrix operand order repaired and checked by reading, not compiled, not run on hardware
//
#include <hip/hip_runtime.h>
#include <hip/hip_bf16.h>

// Problem constants (from reference): B=4, T=12, N=4096, F=16
#define NB   4
#define NT   12
#define NN   4096
#define NF   16
#define C1   512   // stage1 columns: 2 parts * B * 4F
#define C2   384   // stage2 columns: 2 parts * B * 3F
#define KC   128   // K chunk staged in LDS per iteration

typedef __attribute__((ext_vector_type(8)))  float  v8f;
typedef __attribute__((ext_vector_type(16))) __bf16 v16bf;
typedef __attribute__((ext_vector_type(8)))  __bf16 v8bf;
typedef int v4i __attribute__((vector_size(16)));

union AFrag {
    v16bf v;
    struct { v8bf lo; v8bf hi; } p;
};

#if __has_builtin(__builtin_amdgcn_global_load_async_to_lds_b128)
#define HAVE_ASYNC_LDS 1
#else
#define HAVE_ASYNC_LDS 0
#endif

__device__ __forceinline__ float sigm(float v) { return 1.0f / (1.0f + expf(-v)); }

// ---------------------------------------------------------------------------
// Convert fp32 adjacency -> bf16 (read once from HBM, 32MB result lives in L2)
// ---------------------------------------------------------------------------
__global__ __launch_bounds__(256) void cvt_adj(const float* __restrict__ adj,
                                               __bf16* __restrict__ adjb) {
    size_t id = (size_t)blockIdx.x * 256 + threadIdx.x;   // 16M elements
    adjb[id] = (__bf16)adj[id];
}

// ---------------------------------------------------------------------------
// h = c = m = 1.0 initial state
// ---------------------------------------------------------------------------
__global__ __launch_bounds__(256) void init_state(float* h, float* c, float* m) {
    int id = blockIdx.x * 256 + threadIdx.x;              // 262144
    h[id] = 1.0f; c[id] = 1.0f; m[id] = 1.0f;
}

// ---------------------------------------------------------------------------
// Stage-1 projection: P_T[col][n] (bf16, transposed for WMMA-B layout)
//   col = part*256 + b*64 + g ; part0: x_t @ Wx , part1: h @ Wh  (K = F = 16)
// ---------------------------------------------------------------------------
__global__ __launch_bounds__(256) void compute_P1(const float* __restrict__ x,
                                                  const float* __restrict__ h,
                                                  const float* __restrict__ Wx,
                                                  const float* __restrict__ Wh,
                                                  __bf16* __restrict__ Pt, int t) {
    int id  = blockIdx.x * 256 + threadIdx.x;   // C1*NN threads
    int n   = id & (NN - 1);
    int col = id >> 12;
    int part = col >> 8;
    int cp   = col & 255;
    int b    = cp >> 6;
    int g    = cp & 63;
    const float* zrow = part ? (h + ((size_t)b * NN + n) * NF)
                             : (x + ((((size_t)b * NT + t) * NN + n) * NF));
    const float* W = part ? Wh : Wx;            // [F, 64] row-major
    float s = 0.0f;
#pragma unroll
    for (int f = 0; f < NF; ++f) s += zrow[f] * W[f * 64 + g];
    Pt[(size_t)col * NN + n] = (__bf16)s;
}

// ---------------------------------------------------------------------------
// Stage-2 projection: col = part*192 + b*48 + g ; part0: h @ Wsh, part1: m @ Wsm
// ---------------------------------------------------------------------------
__global__ __launch_bounds__(256) void compute_P2(const float* __restrict__ h,
                                                  const float* __restrict__ m,
                                                  const float* __restrict__ Wsh,
                                                  const float* __restrict__ Wsm,
                                                  __bf16* __restrict__ Pt) {
    int id  = blockIdx.x * 256 + threadIdx.x;   // C2*NN threads
    int n   = id & (NN - 1);
    int col = id >> 12;                         // 0..383
    int part = (col >= 192) ? 1 : 0;
    int cp   = col - part * 192;
    int b    = cp / 48;
    int g    = cp % 48;
    const float* zrow = (part ? m : h) + ((size_t)b * NN + n) * NF;
    const float* W = part ? Wsm : Wsh;          // [F, 48] row-major
    float s = 0.0f;
#pragma unroll
    for (int f = 0; f < NF; ++f) s += zrow[f] * W[f * 48 + g];
    Pt[(size_t)col * NN + n] = (__bf16)s;
}

// ---------------------------------------------------------------------------
// Core GEMM: G[m][col] = relu( adj[m,:] @ P_T[col,:]^T + bias[part][g] )
// bf16 A/B, fp32 accumulate via v_wmma_f32_16x16x32_bf16.
// Block = 8 wave32 arranged 2x4; block tile = 64 rows x 128 cols; each wave
// computes a 32x32 macro-tile (2x2 WMMA tiles) so every A/B fragment feeds
// two WMMAs. adj strip (64 x KC bf16, 16KB) staged in LDS via async-to-LDS
// (GLOBAL_LOAD_ASYNC_TO_LDS_B128, ASYNCcnt-tracked). Grid = (N/64, C/128).
// ---------------------------------------------------------------------------
__global__ __launch_bounds__(256) void gconv_gemm(const __bf16* __restrict__ adjb,
                                                  const __bf16* __restrict__ Pt,
                                                  const float* __restrict__ bias0,
                                                  const float* __restrict__ bias1,
                                                  float* __restrict__ G,
                                                  int C, int partCols, int gates) {
    __shared__ __attribute__((aligned(16))) __bf16 ldsA[64 * KC];   // 16KB strip

    const int tid  = threadIdx.x;
    const int lane = tid & 31;
    const int wave = tid >> 5;
    const int lm   = lane & 15;     // A: row-in-tile, B/D: col-in-tile
    const int half = lane >> 4;
    const int rg   = wave >> 2;     // 0..1 : row group (32 rows each)
    const int cg   = wave & 3;      // 0..3 : col group (32 cols each)
    const int m0   = blockIdx.x * 64;
    const int colBase = blockIdx.y * 128 + cg * 32;

    v8f acc[2][2] = {};

    for (int k0 = 0; k0 < NN; k0 += KC) {
        // stage adj[m0:m0+64, k0:k0+KC] into LDS: 1024 chunks of 16B, 4/thread
#pragma unroll
        for (int u = 0; u < 4; ++u) {
            int e  = tid + u * 256;     // 0..1023
            int r  = e >> 4;            // row 0..63 (16 chunks per row)
            int ch = e & 15;
            const __bf16* src = adjb + (size_t)(m0 + r) * NN + k0 + ch * 8;
            __bf16* dst = &ldsA[r * KC + ch * 8];
#if HAVE_ASYNC_LDS
            __builtin_amdgcn_global_load_async_to_lds_b128(
                (__attribute__((address_space(1))) v4i*)src,
                (__attribute__((address_space(3))) v4i*)dst, 0, 0);
#else
            *(v8bf*)dst = *(const v8bf*)src;
#endif
        }
#if HAVE_ASYNC_LDS
#if __has_builtin(__builtin_amdgcn_s_wait_asynccnt)
        __builtin_amdgcn_s_wait_asynccnt(0);
#else
        asm volatile("s_wait_asynccnt 0x0" ::: "memory");
#endif
#endif
        // hint the next adj strip chunk toward the caches (global_prefetch_b8)
        if (k0 + KC < NN)
            __builtin_prefetch(adjb + (size_t)(m0 + (tid & 63)) * NN + k0 + KC, 0, 0);
        __syncthreads();

#pragma unroll
        for (int kk = 0; kk < KC; kk += 32) {
            // A fragments (16x32 bf16) for the two row tiles
            AFrag a0, a1;
            const __bf16* ar0 = &ldsA[(rg * 32 + lm) * KC + kk + half * 16];
            a0.p.lo = *(const v8bf*)(ar0);
            a0.p.hi = *(const v8bf*)(ar0 + 8);
            const __bf16* ar1 = ar0 + 16 * KC;
            a1.p.lo = *(const v8bf*)(ar1);
            a1.p.hi = *(const v8bf*)(ar1 + 8);
            // B fragments (32x16 bf16) for the two col tiles: 32B contiguous
            const __bf16* br0 = Pt + (size_t)(colBase + lm) * NN + k0 + kk + half * 16;
            v16bf b0 = *(const v16bf*)(br0);
            v16bf b1 = *(const v16bf*)(br0 + 16 * NN);
            acc[0][0] = __builtin_amdgcn_wmma_f32_16x16x32_bf16(
                false, a0.v, false, b0, (short)0, acc[0][0], false, false);
            acc[0][1] = __builtin_amdgcn_wmma_f32_16x16x32_bf16(
                false, a0.v, false, b1, (short)0, acc[0][1], false, false);
            acc[1][0] = __builtin_amdgcn_wmma_f32_16x16x32_bf16(
                false, a1.v, false, b0, (short)0, acc[1][0], false, false);
            acc[1][1] = __builtin_amdgcn_wmma_f32_16x16x32_bf16(
                false, a1.v, false, b1, (short)0, acc[1][1], false, false);
        }
        __syncthreads();
    }

    // epilogue: bias + relu, store fp32 G row-major [N][C]
#pragma unroll
    for (int j = 0; j < 2; ++j) {
#pragma unroll
        for (int i = 0; i < 2; ++i) {
            int col  = colBase + i * 16 + lm;
            int part = (col >= partCols) ? 1 : 0;
            int cp   = col - part * partCols;
            float bias = part ? bias1[cp % gates] : bias0[cp % gates];
#pragma unroll
            for (int r = 0; r < 8; ++r) {
                int row = m0 + rg * 32 + j * 16 + half * 8 + r;
                float v = acc[j][i][r] + bias;
                G[(size_t)row * C + col] = v > 0.0f ? v : 0.0f;
            }
        }
    }
}

// ---------------------------------------------------------------------------
// LSTM gate update: fg,ig,og,c,h from G1 [N][512]
// ---------------------------------------------------------------------------
__global__ __launch_bounds__(256) void lstm_gates(const float* __restrict__ G1,
                                                  float* __restrict__ h,
                                                  float* __restrict__ c) {
    int id = blockIdx.x * 256 + threadIdx.x;    // B*N*F = 262144
    int b  = id >> 16;
    int n  = (id >> 4) & (NN - 1);
    int f  = id & 15;
    const float* Gr = G1 + (size_t)n * C1;
    int cb = b * 64 + f;
    float fx = Gr[cb],        ix = Gr[cb + 16],        cx = Gr[cb + 32],        ox = Gr[cb + 48];
    float fh = Gr[256 + cb],  ih = Gr[256 + cb + 16],  chv = Gr[256 + cb + 32], oh = Gr[256 + cb + 48];
    float fg = sigm(fx + fh);
    float ig = sigm(ix + ih);
    float og = sigm(ox + oh);
    float cn = fg * c[id] + ig * tanhf(cx + chv);
    c[id] = cn;
    h[id] = og * tanhf(cn);
}

// ---------------------------------------------------------------------------
// Memory-stage update from G2 [N][384]; also writes hidden_states[t]
// ---------------------------------------------------------------------------
__global__ __launch_bounds__(256) void mem_update(const float* __restrict__ G2,
                                                  float* __restrict__ h,
                                                  float* __restrict__ m,
                                                  float* __restrict__ hs, int t) {
    int id = blockIdx.x * 256 + threadIdx.x;    // 262144
    int b  = id >> 16;
    int n  = (id >> 4) & (NN - 1);
    int f  = id & 15;
    const float* Gr = G2 + (size_t)n * C2;
    int cb = b * 48 + f;
    float i_h = Gr[cb],       g_h = Gr[cb + 16],       o_h = Gr[cb + 32];
    float i_m = Gr[192 + cb], g_m = Gr[192 + cb + 16], o_m = Gr[192 + cb + 32];
    float i2 = sigm(i_h + i_m);
    float g2 = sigm(g_h + g_m);
    float o2 = sigm(o_h + o_m);
    float mn = i2 * m[id] + (1.0f - i2) * g2;
    m[id] = mn;
    float hn = mn * o2;
    h[id] = hn;
    hs[(((size_t)b * NT + t) * NN + n) * NF + f] = hn;
}

// ---------------------------------------------------------------------------
extern "C" void kernel_launch(void* const* d_in, const int* in_sizes, int n_in,
                              void* d_out, int out_size, void* d_ws, size_t ws_size,
                              hipStream_t stream) {
    const float* x   = (const float*)d_in[0];
    const float* adj = (const float*)d_in[1];
    const float* Wx  = (const float*)d_in[2];
    const float* bx  = (const float*)d_in[3];
    const float* Wh  = (const float*)d_in[4];
    const float* bh  = (const float*)d_in[5];
    const float* Wsh = (const float*)d_in[6];
    const float* bsh = (const float*)d_in[7];
    const float* Wsm = (const float*)d_in[8];
    const float* bsm = (const float*)d_in[9];
    float* out = (float*)d_out;

    char* ws = (char*)d_ws;
    __bf16* adjb = (__bf16*)ws;                        // 32 MB
    __bf16* Pt   = (__bf16*)(ws + 33554432);           // 4 MB (C1 x N bf16)
    float*  G    = (float*)(ws + 37748736);            // 8 MB (N x C1 f32)
    float*  h    = (float*)(ws + 46137344);            // 1 MB
    float*  c    = (float*)(ws + 47185920);            // 1 MB
    float*  m    = (float*)(ws + 48234496);            // 1 MB

    cvt_adj<<<(NN * NN) / 256, 256, 0, stream>>>(adj, adjb);
    init_state<<<(NB * NN * NF) / 256, 256, 0, stream>>>(h, c, m);

    for (int t = 0; t < NT; ++t) {
        compute_P1<<<(C1 * NN) / 256, 256, 0, stream>>>(x, h, Wx, Wh, Pt, t);
        gconv_gemm<<<dim3(NN / 64, C1 / 128), 256, 0, stream>>>(
            adjb, Pt, bx, bh, G, C1, 256, 64);
        lstm_gates<<<(NB * NN * NF) / 256, 256, 0, stream>>>(G, h, c);
        compute_P2<<<(C2 * NN) / 256, 256, 0, stream>>>(h, m, Wsh, Wsm, Pt);
        gconv_gemm<<<dim3(NN / 64, C2 / 128), 256, 0, stream>>>(
            adjb, Pt, bsh, bsm, G, C2, 192, 48);
        mem_update<<<(NB * NN * NF) / 256, 256, 0, stream>>>(G, h, m, out, t);
    }

    size_t tail = (size_t)NB * NT * NN * NF;
    size_t snf  = (size_t)NB * NN * NF;
    (void)hipMemcpyAsync(out + tail,           h, snf * sizeof(float), hipMemcpyDeviceToDevice, stream);
    (void)hipMemcpyAsync(out + tail + snf,     c, snf * sizeof(float), hipMemcpyDeviceToDevice, stream);
    (void)hipMemcpyAsync(out + tail + 2 * snf, m, snf * sizeof(float), hipMemcpyDeviceToDevice, stream);
}
